// ParallelMLPs_32435593019746
// MI455X (gfx1250) — compile-verified, matrix operand order checked by reading, not run on hardware
//
#include <hip/hip_runtime.h>

// ---------------------------------------------------------------------------
// ParallelMLPs: out[b] = silu(x[b] @ W1[b] + b1[b]) @ W2[b] + b2[b]
// B=8, S=2048, D=1024, H=4096, fp32 in/out.
// Compute-bound (275 GFLOP vs ~400MB minimum traffic @23.3TB/s ~= 17us) ->
// run both GEMMs as BF16 WMMA with f32 accumulation, h kept in bf16 in d_ws.
// bf16 pairs converted with the packed cvt (v_cvt_pk_bf16_f32), B tiles staged
// transposed with packed b32 DS stores, next K-tile prefetched during WMMAs.
// ---------------------------------------------------------------------------

typedef __attribute__((ext_vector_type(16))) __bf16        bf16x16;
typedef __attribute__((ext_vector_type(2)))  __bf16        bf16x2;
typedef __attribute__((ext_vector_type(8)))  float         f32x8;
typedef __attribute__((ext_vector_type(4)))  float         f32x4;
typedef __attribute__((ext_vector_type(2)))  float         f32x2;
typedef __attribute__((ext_vector_type(4)))  unsigned int  u32x4;

#define BATCH 8
#define SEQ   2048
#define DIM   1024
#define HID   4096

#define BM 128
#define BN 128
#define BK 32
#define LDT 40   // padded LDS row stride (bf16 elems): 80B -> conflict-free frag loads

// Pack two floats into two bf16 with ONE packed convert where available.
__device__ __forceinline__ unsigned int pk_bf16(float a, float b) {
#if __has_builtin(__builtin_amdgcn_cvt_pk_bf16_f32)
  union { bf16x2 h; unsigned int u; } r;
  r.h = __builtin_amdgcn_cvt_pk_bf16_f32(a, b);
  return r.u;
#else
  union { bf16x2 h; unsigned int u; } r;
  r.h = __builtin_convertvector((f32x2){a, b}, bf16x2);  // -> v_cvt_pk_bf16_f32
  return r.u;
#endif
}

__device__ __forceinline__ unsigned short one_bf16(float a) {
  union { __bf16 h; unsigned short u; } r;
  r.h = (__bf16)a;
  return r.u;
}

union FragBF16 {
  u32x4   q[2];
  bf16x16 v;
};

// A/B fragment load matching the 16-bit 16x32 WMMA operand layout:
// lane m (m=lane&15, hi=lane>>4) holds k in [8*hi, 8*hi+8) and [16+8*hi, ...)
// = two contiguous 16B chunks per lane from an LDS row.
__device__ __forceinline__ void load_frag(const unsigned short* lds,
                                          int rowBase, int lane, FragBF16& f) {
  const int m  = lane & 15;
  const int hi = lane >> 4;
  const unsigned short* p = lds + (rowBase + m) * LDT + 8 * hi;
  f.q[0] = *(const u32x4*)(p);        // k: 8*hi .. 8*hi+7
  f.q[1] = *(const u32x4*)(p + 16);   // k: 16+8*hi .. 16+8*hi+7
}

// ---------------------------------------------------------------------------
// Kernel 1: h[b] = silu(x[b] (S x D) @ W1[b] (D x H) + b1[b]), h stored bf16
// ---------------------------------------------------------------------------
__global__ __launch_bounds__(256)
void mlp_gemm1_silu(const float* __restrict__ x, const float* __restrict__ W1,
                    const float* __restrict__ b1, unsigned short* __restrict__ h) {
  __shared__ unsigned short As[BM * LDT];   // x tile,  [row][k]
  __shared__ unsigned short Bs[BN * LDT];   // W1 tile, transposed: [col][k]

  const int b    = blockIdx.z;
  const int m0   = blockIdx.y * BM;
  const int n0   = blockIdx.x * BN;
  const int tid  = threadIdx.x;
  const int lane = tid & 31;
  const int wave = tid >> 5;
  const int wm   = wave & 3;   // 4 waves over M: 32 rows each
  const int wn   = wave >> 2;  // 2 waves over N: 64 cols each

  const float* xb  = x  + (size_t)b * SEQ * DIM;
  const float* W1b = W1 + (size_t)b * DIM * HID;

  f32x8 acc[2][4];
  #pragma unroll
  for (int i = 0; i < 2; ++i)
    #pragma unroll
    for (int j = 0; j < 4; ++j)
      acc[i][j] = (f32x8){0.f, 0.f, 0.f, 0.f, 0.f, 0.f, 0.f, 0.f};

  // per-thread fixed staging coordinates
  const int arow = tid >> 3;              // A: 8 float4-chunks per 32-wide row
  const int akc  = (tid & 7) << 2;
  const int bkp  = tid >> 5;              // B: k-pair index base (0..7 for i=0)
  const int bnc  = (tid & 31) << 2;       // B: 4-col chunk

  for (int k0 = 0; k0 < DIM; k0 += BK) {
    // ---- stage A: 128x32 f32 -> bf16 LDS (coalesced float4 loads) ----
    #pragma unroll
    for (int i = 0; i < 4; ++i) {
      const int row = arow + i * 32;
      const f32x4 v = *(const f32x4*)(xb + (size_t)(m0 + row) * DIM + k0 + akc);
      const unsigned int p0 = pk_bf16(v.x, v.y);
      const unsigned int p1 = pk_bf16(v.z, v.w);
      *(unsigned long long*)(As + row * LDT + akc) =
          ((unsigned long long)p1 << 32) | (unsigned long long)p0;
    }
    // ---- stage B: 32x128 f32, two k-rows per thread, packed b32 transpose ----
    #pragma unroll
    for (int i = 0; i < 2; ++i) {
      const int kp = bkp + i * 8;         // k-pair: rows 2kp, 2kp+1
      const float* p = W1b + (size_t)(k0 + 2 * kp) * HID + n0 + bnc;
      const f32x4 r0 = *(const f32x4*)(p);
      const f32x4 r1 = *(const f32x4*)(p + HID);
      *(unsigned int*)(Bs + (bnc + 0) * LDT + 2 * kp) = pk_bf16(r0.x, r1.x);
      *(unsigned int*)(Bs + (bnc + 1) * LDT + 2 * kp) = pk_bf16(r0.y, r1.y);
      *(unsigned int*)(Bs + (bnc + 2) * LDT + 2 * kp) = pk_bf16(r0.z, r1.z);
      *(unsigned int*)(Bs + (bnc + 3) * LDT + 2 * kp) = pk_bf16(r0.w, r1.w);
    }
    // prefetch next K-tile while this one is computed (global_prefetch_b8)
    if (k0 + BK < DIM) {
      __builtin_prefetch(xb + (size_t)(m0 + arow) * DIM + k0 + BK + akc, 0, 1);
      __builtin_prefetch(W1b + (size_t)(k0 + BK + 2 * bkp) * HID + n0 + bnc, 0, 1);
    }
    __syncthreads();

    FragBF16 af[2], bf[4];
    #pragma unroll
    for (int i = 0; i < 2; ++i) load_frag(As, wm * 32 + i * 16, lane, af[i]);
    #pragma unroll
    for (int j = 0; j < 4; ++j) load_frag(Bs, wn * 64 + j * 16, lane, bf[j]);
    #pragma unroll
    for (int i = 0; i < 2; ++i)
      #pragma unroll
      for (int j = 0; j < 4; ++j)
        acc[i][j] = __builtin_amdgcn_wmma_f32_16x16x32_bf16(
            false, af[i].v, false, bf[j].v, (short)0, acc[i][j], false, false);
    __syncthreads();
  }

  // ---- epilogue: bias + SiLU, store bf16 h ----
  unsigned short* hb = h + (size_t)b * SEQ * HID;
  const int nl = lane & 15;
  const int rs = (lane >> 4) * 8;    // C/D layout: lanes 16-31 hold rows 8..15
  #pragma unroll
  for (int i = 0; i < 2; ++i) {
    const int rowBase = m0 + wm * 32 + i * 16 + rs;
    #pragma unroll
    for (int j = 0; j < 4; ++j) {
      const int col  = n0 + wn * 64 + j * 16 + nl;
      const float bv = b1[b * HID + col];
      #pragma unroll
      for (int r = 0; r < 8; ++r) {
        const float t = acc[i][j][r] + bv;
        const float s = t * (1.0f / (1.0f + __expf(-t)));   // silu
        hb[(size_t)(rowBase + r) * HID + col] = one_bf16(s);
      }
    }
  }
}

// ---------------------------------------------------------------------------
// Kernel 2: out[b] = h[b] (S x H, bf16) @ W2[b] (H x D) + b2[b], fp32 out
// ---------------------------------------------------------------------------
__global__ __launch_bounds__(256)
void mlp_gemm2(const unsigned short* __restrict__ h, const float* __restrict__ W2,
               const float* __restrict__ b2, float* __restrict__ out) {
  __shared__ unsigned short As[BM * LDT];
  __shared__ unsigned short Bs[BN * LDT];

  const int b    = blockIdx.z;
  const int m0   = blockIdx.y * BM;
  const int n0   = blockIdx.x * BN;   // over DIM
  const int tid  = threadIdx.x;
  const int lane = tid & 31;
  const int wave = tid >> 5;
  const int wm   = wave & 3;
  const int wn   = wave >> 2;

  const unsigned short* hb  = h  + (size_t)b * SEQ * HID;
  const float*          W2b = W2 + (size_t)b * HID * DIM;

  f32x8 acc[2][4];
  #pragma unroll
  for (int i = 0; i < 2; ++i)
    #pragma unroll
    for (int j = 0; j < 4; ++j)
      acc[i][j] = (f32x8){0.f, 0.f, 0.f, 0.f, 0.f, 0.f, 0.f, 0.f};

  const int arow = tid >> 2;              // A: 4 x 8-elem chunks per 32-wide row
  const int akc  = (tid & 3) << 3;
  const int bkp  = tid >> 5;
  const int bnc  = (tid & 31) << 2;

  for (int k0 = 0; k0 < HID; k0 += BK) {
    // ---- stage A: 128x32 bf16 tile of h, straight 16B copies ----
    #pragma unroll
    for (int i = 0; i < 2; ++i) {
      const int row = arow + i * 64;
      const u32x4 v = *(const u32x4*)(hb + (size_t)(m0 + row) * HID + k0 + akc);
      *(u32x4*)(As + row * LDT + akc) = v;
    }
    // ---- stage B: 32x128 f32 W2 tile, packed b32 transpose ----
    #pragma unroll
    for (int i = 0; i < 2; ++i) {
      const int kp = bkp + i * 8;
      const float* p = W2b + (size_t)(k0 + 2 * kp) * DIM + n0 + bnc;
      const f32x4 r0 = *(const f32x4*)(p);
      const f32x4 r1 = *(const f32x4*)(p + DIM);
      *(unsigned int*)(Bs + (bnc + 0) * LDT + 2 * kp) = pk_bf16(r0.x, r1.x);
      *(unsigned int*)(Bs + (bnc + 1) * LDT + 2 * kp) = pk_bf16(r0.y, r1.y);
      *(unsigned int*)(Bs + (bnc + 2) * LDT + 2 * kp) = pk_bf16(r0.z, r1.z);
      *(unsigned int*)(Bs + (bnc + 3) * LDT + 2 * kp) = pk_bf16(r0.w, r1.w);
    }
    if (k0 + BK < HID) {
      __builtin_prefetch(hb + (size_t)(m0 + arow) * HID + k0 + BK + akc, 0, 1);
      __builtin_prefetch(W2b + (size_t)(k0 + BK + 2 * bkp) * DIM + n0 + bnc, 0, 1);
    }
    __syncthreads();

    FragBF16 af[2], bf[4];
    #pragma unroll
    for (int i = 0; i < 2; ++i) load_frag(As, wm * 32 + i * 16, lane, af[i]);
    #pragma unroll
    for (int j = 0; j < 4; ++j) load_frag(Bs, wn * 64 + j * 16, lane, bf[j]);
    #pragma unroll
    for (int i = 0; i < 2; ++i)
      #pragma unroll
      for (int j = 0; j < 4; ++j)
        acc[i][j] = __builtin_amdgcn_wmma_f32_16x16x32_bf16(
            false, af[i].v, false, bf[j].v, (short)0, acc[i][j], false, false);
    __syncthreads();
  }

  // ---- epilogue: bias, fp32 store ----
  float* ob = out + (size_t)b * SEQ * DIM;
  const int nl = lane & 15;
  const int rs = (lane >> 4) * 8;
  #pragma unroll
  for (int i = 0; i < 2; ++i) {
    const int rowBase = m0 + wm * 32 + i * 16 + rs;
    #pragma unroll
    for (int j = 0; j < 4; ++j) {
      const int col  = n0 + wn * 64 + j * 16 + nl;
      const float bv = b2[b * DIM + col];
      #pragma unroll
      for (int r = 0; r < 8; ++r)
        ob[(size_t)(rowBase + r) * DIM + col] = acc[i][j][r] + bv;
    }
  }
}

// ---------------------------------------------------------------------------
extern "C" void kernel_launch(void* const* d_in, const int* in_sizes, int n_in,
                              void* d_out, int out_size, void* d_ws, size_t ws_size,
                              hipStream_t stream) {
  const float* x  = (const float*)d_in[0];  // [B,S,D]
  const float* W1 = (const float*)d_in[1];  // [B,D,H]
  const float* b1 = (const float*)d_in[2];  // [B,H]
  const float* W2 = (const float*)d_in[3];  // [B,H,D]
  const float* b2 = (const float*)d_in[4];  // [B,D]
  float* out = (float*)d_out;               // [B,S,D] fp32
  unsigned short* h = (unsigned short*)d_ws; // [B,S,H] bf16 intermediate (128 MiB)

  dim3 block(256, 1, 1);
  dim3 g1(HID / BN, SEQ / BM, BATCH);  // 32 x 16 x 8
  dim3 g2(DIM / BN, SEQ / BM, BATCH);  //  8 x 16 x 8

  mlp_gemm1_silu<<<g1, block, 0, stream>>>(x, W1, b1, h);
  mlp_gemm2<<<g2, block, 0, stream>>>(h, W2, b2, out);
}
